// GlobalHypergraph_29506425324179
// MI455X (gfx1250) — matrix operand order repaired and checked by reading, not compiled, not run on hardware
//
#include <hip/hip_runtime.h>
#include <math.h>

// GlobalHypergraph pipeline for MI455X (gfx1250), wave32 + WMMA f16->f32.
// Round 3: f16-resident GEMM operands (weights/activations pre-converted),
// fragment loads are raw b128s matching the WMMA register layout, and the
// K-loop is software-pipelined (prefetch next A + 4xB before the 4 WMMAs).

typedef __attribute__((ext_vector_type(16))) _Float16 v16h;
typedef __attribute__((ext_vector_type(8)))  _Float16 v8h;
typedef __attribute__((ext_vector_type(8)))  float    v8f;

#define GH_EPS 1e-6f

#define GH_WMMA(acc, a, b) \
    acc = __builtin_amdgcn_wmma_f32_16x16x32_f16(false, a, false, b, (short)0, acc, false, false)

// A fragment (ISA 7.12.2): lane m=l16, k = (e<8?e:e+8)+8*half -> two 8-half runs
__device__ __forceinline__ v16h gh_ldA16(const _Float16* p)
{
    v8h lo = *(const v8h*)(p);
    v8h hi = *(const v8h*)(p + 16);
    return __builtin_shufflevector(lo, hi, 0,1,2,3,4,5,6,7,8,9,10,11,12,13,14,15);
}
// B fragment: lane n=l16, k = (e<8?e:e+8)+8*half -> two 8-half runs (same as A)
__device__ __forceinline__ v16h gh_ldB16(const _Float16* p) { return gh_ldA16(p); }

__device__ __forceinline__ float4 gh_ld4(const float* p) { return *(const float4*)p; }

template<int O>
__device__ __forceinline__ void gh_cvt8(v16h& d, float4 a, float4 b)
{
    d[O + 0] = (_Float16)a.x; d[O + 1] = (_Float16)a.y;
    d[O + 2] = (_Float16)a.z; d[O + 3] = (_Float16)a.w;
    d[O + 4] = (_Float16)b.x; d[O + 5] = (_Float16)b.y;
    d[O + 6] = (_Float16)b.z; d[O + 7] = (_Float16)b.w;
}
__device__ __forceinline__ float4 gh_mul4(float4 a, float4 c)
{
    a.x *= c.x; a.y *= c.y; a.z *= c.z; a.w *= c.w; return a;
}

// ---------------------------------------------------------------------------
// f16-operand GEMM:  C = act(scale * A@B.T + bias)
// A[M,K] f16 row-major, B[N,K] f16 row-major (computing @B.T), C f16 or f32.
// 16x64 strip per wave32; grid = (N/64, M/16); K%32==0, K>=64.
// Software-pipelined: next k-step's 5 fragments load while 4 WMMAs issue.
// ---------------------------------------------------------------------------
template<bool BIAS, int ACT, bool CH>
__global__ __launch_bounds__(32)
void gh_gemm_h(const _Float16* __restrict__ A, const _Float16* __restrict__ B,
               const float* __restrict__ bias, void* __restrict__ Cv,
               int K, int lda, int ldb, int ldc, float scale)
{
    const int lane = threadIdx.x & 31;
    const int half = lane >> 4;
    const int l16  = lane & 15;
    const int tile_n = blockIdx.x << 6;
    const int tile_m = blockIdx.y << 4;

    const _Float16* arow = A + (size_t)(tile_m + l16) * lda + (half << 3);
    const _Float16* bp0  = B + (size_t)(tile_n + l16) * ldb + (half << 3);
    const _Float16* bp1  = bp0 + (size_t)16 * ldb;
    const _Float16* bp2  = bp0 + (size_t)32 * ldb;
    const _Float16* bp3  = bp0 + (size_t)48 * ldb;

    v8f ac0 = (v8f){0.f,0.f,0.f,0.f,0.f,0.f,0.f,0.f};
    v8f ac1 = ac0, ac2 = ac0, ac3 = ac0;

    v16h af = gh_ldA16(arow);
    v16h b0 = gh_ldB16(bp0), b1 = gh_ldB16(bp1);
    v16h b2 = gh_ldB16(bp2), b3 = gh_ldB16(bp3);

    for (int k0 = 32; k0 < K; k0 += 32) {
        v16h afn = gh_ldA16(arow + k0);
        v16h b0n = gh_ldB16(bp0 + k0);
        v16h b1n = gh_ldB16(bp1 + k0);
        v16h b2n = gh_ldB16(bp2 + k0);
        v16h b3n = gh_ldB16(bp3 + k0);
        GH_WMMA(ac0, af, b0); GH_WMMA(ac1, af, b1);
        GH_WMMA(ac2, af, b2); GH_WMMA(ac3, af, b3);
        af = afn; b0 = b0n; b1 = b1n; b2 = b2n; b3 = b3n;
    }
    GH_WMMA(ac0, af, b0); GH_WMMA(ac1, af, b1);
    GH_WMMA(ac2, af, b2); GH_WMMA(ac3, af, b3);

    v8f accs[4] = {ac0, ac1, ac2, ac3};
#pragma unroll
    for (int t = 0; t < 4; ++t) {
        const int n = tile_n + (t << 4) + l16;
        float bv = 0.f;
        if constexpr (BIAS) bv = bias[n];
#pragma unroll
        for (int v = 0; v < 8; ++v) {
            const int m = tile_m + v + (half << 3);
            float r = accs[t][v] * scale;
            if constexpr (BIAS) r += bv;
            if constexpr (ACT == 1) r = fmaxf(r, 0.f);
            if constexpr (ACT == 2) r = (r > 0.f) ? r : (__expf(r) - 1.f);
            if constexpr (CH) ((_Float16*)Cv)[(size_t)m * ldc + n] = (_Float16)r;
            else              ((float*)Cv)[(size_t)m * ldc + n] = r;
        }
    }
}

// ---------------------------------------------------------------------------
// f32-operand GEMM for the transposed-access hypergraph-conv tail.
//   TA=0: a(m,k)=A[m*lda+k]   TA=1: a(m,k)=A[k*lda+m]
//   TB=1: b(k,n)=B[k*ldb+n]   CS: a *= colscale[k]   RES: += residual
// ---------------------------------------------------------------------------
template<int TA, bool CS, bool RES, int ACT>
__global__ __launch_bounds__(32)
void gh_gemm_f(const float* __restrict__ A, const float* __restrict__ B,
               const float* __restrict__ colscale, const float* __restrict__ residual,
               float* __restrict__ C, int K, int lda, int ldb, int ldc, float scale)
{
    const int lane = threadIdx.x & 31;
    const int half = lane >> 4;
    const int l16  = lane & 15;
    const int tile_n = blockIdx.x << 6;
    const int tile_m = blockIdx.y << 4;
    const int am = tile_m + l16;

    v8f acc[4];
#pragma unroll
    for (int t = 0; t < 4; ++t) acc[t] = (v8f){0.f,0.f,0.f,0.f,0.f,0.f,0.f,0.f};

    for (int k0 = 0; k0 < K; k0 += 32) {
        v16h af;
        if constexpr (TA == 0) {
            const float* ap = A + (size_t)am * lda + k0 + (half << 3);
            float4 a0 = gh_ld4(ap +  0), a1 = gh_ld4(ap +  4);
            float4 a2 = gh_ld4(ap + 16), a3 = gh_ld4(ap + 20);
            if constexpr (CS) {
                const float* cp = colscale + k0 + (half << 3);
                a0 = gh_mul4(a0, gh_ld4(cp +  0));
                a1 = gh_mul4(a1, gh_ld4(cp +  4));
                a2 = gh_mul4(a2, gh_ld4(cp + 16));
                a3 = gh_mul4(a3, gh_ld4(cp + 20));
            }
            gh_cvt8<0>(af, a0, a1);
            gh_cvt8<8>(af, a2, a3);
        } else {
#pragma unroll
            for (int e = 0; e < 16; ++e) {
                int ka = k0 + ((e < 8) ? e : (e + 8)) + (half << 3);
                af[e] = (_Float16)A[(size_t)ka * lda + am];
            }
        }
#pragma unroll
        for (int t = 0; t < 4; ++t) {
            const int bn = tile_n + (t << 4) + l16;
            v16h bf;
#pragma unroll
            for (int e = 0; e < 16; ++e) {
                int kb = k0 + ((e < 8) ? e : (e + 8)) + (half << 3);
                bf[e] = (_Float16)B[(size_t)kb * ldb + bn];
            }
            GH_WMMA(acc[t], af, bf);
        }
    }
#pragma unroll
    for (int t = 0; t < 4; ++t) {
        const int n = tile_n + (t << 4) + l16;
#pragma unroll
        for (int v = 0; v < 8; ++v) {
            const int m = tile_m + v + (half << 3);
            float r = acc[t][v] * scale;
            if constexpr (ACT == 1) r = fmaxf(r, 0.f);
            if constexpr (ACT == 2) r = (r > 0.f) ? r : (__expf(r) - 1.f);
            if constexpr (RES) r += residual[(size_t)m * ldc + n];
            C[(size_t)m * ldc + n] = r;
        }
    }
}

// f32 -> f16 conversion (n % 1024 == 0)
__global__ __launch_bounds__(256)
void gh_tof16(const float* __restrict__ src, _Float16* __restrict__ dst, int n)
{
    int i = (blockIdx.x * 256 + threadIdx.x) * 4;
    if (i + 3 < n) {
        float4 v = gh_ld4(src + i);
        dst[i + 0] = (_Float16)v.x; dst[i + 1] = (_Float16)v.y;
        dst[i + 2] = (_Float16)v.z; dst[i + 3] = (_Float16)v.w;
    }
}

// a[row,j] = 0.5*(softmax(s[0,row,:])[j] + softmax(s[1,row,:])[j]) -> f16
__global__ __launch_bounds__(256)
void gh_softmax_rows_avg(const float* __restrict__ s, _Float16* __restrict__ a)
{
    const int row = blockIdx.x;
    const int t   = threadIdx.x;
    __shared__ float sm[256];
    float outv[4] = {0.f, 0.f, 0.f, 0.f};

    for (int h = 0; h < 2; ++h) {
        const float* sr = s + (size_t)h * 1024 * 1024 + (size_t)row * 1024;
        float x[4];
        float mx = -3.4e38f;
#pragma unroll
        for (int u = 0; u < 4; ++u) { x[u] = sr[t + (u << 8)]; mx = fmaxf(mx, x[u]); }
        sm[t] = mx; __syncthreads();
        for (int st = 128; st > 0; st >>= 1) { if (t < st) sm[t] = fmaxf(sm[t], sm[t + st]); __syncthreads(); }
        mx = sm[0]; __syncthreads();
        float ls = 0.f;
#pragma unroll
        for (int u = 0; u < 4; ++u) { x[u] = __expf(x[u] - mx); ls += x[u]; }
        sm[t] = ls; __syncthreads();
        for (int st = 128; st > 0; st >>= 1) { if (t < st) sm[t] += sm[t + st]; __syncthreads(); }
        float inv = 0.5f / sm[0]; __syncthreads();
#pragma unroll
        for (int u = 0; u < 4; ++u) outv[u] += x[u] * inv;
    }
    _Float16* ar = a + (size_t)row * 1024;
#pragma unroll
    for (int u = 0; u < 4; ++u) ar[t + (u << 8)] = (_Float16)outv[u];
}

// column standardize (ddof=1 over 1024) + softmax over rows; out f16 or f32
template<bool OH>
__global__ __launch_bounds__(256)
void gh_norm_softmax_col(const float* __restrict__ X, int ldx,
                         void* __restrict__ Yv, int ldy, int ycol0)
{
    const int col = blockIdx.x;
    const int t   = threadIdx.x;
    __shared__ float sm[256];
    float x[4];
    float s = 0.f;
#pragma unroll
    for (int u = 0; u < 4; ++u) { x[u] = X[(size_t)(t + (u << 8)) * ldx + col]; s += x[u]; }
    sm[t] = s; __syncthreads();
    for (int st = 128; st > 0; st >>= 1) { if (t < st) sm[t] += sm[t + st]; __syncthreads(); }
    const float mean = sm[0] * (1.0f / 1024.0f); __syncthreads();

    float ss = 0.f;
#pragma unroll
    for (int u = 0; u < 4; ++u) { float d = x[u] - mean; ss += d * d; }
    sm[t] = ss; __syncthreads();
    for (int st = 128; st > 0; st >>= 1) { if (t < st) sm[t] += sm[t + st]; __syncthreads(); }
    const float sd = sqrtf(sm[0] * (1.0f / 1023.0f)) + GH_EPS; __syncthreads();

    float mx = -3.4e38f;
#pragma unroll
    for (int u = 0; u < 4; ++u) { x[u] = (x[u] - mean) / sd; mx = fmaxf(mx, x[u]); }
    sm[t] = mx; __syncthreads();
    for (int st = 128; st > 0; st >>= 1) { if (t < st) sm[t] = fmaxf(sm[t], sm[t + st]); __syncthreads(); }
    mx = sm[0]; __syncthreads();

    float es = 0.f;
#pragma unroll
    for (int u = 0; u < 4; ++u) { x[u] = __expf(x[u] - mx); es += x[u]; }
    sm[t] = es; __syncthreads();
    for (int st = 128; st > 0; st >>= 1) { if (t < st) sm[t] += sm[t + st]; __syncthreads(); }
    const float inv = 1.0f / sm[0];
#pragma unroll
    for (int u = 0; u < 4; ++u) {
        const size_t idx = (size_t)(t + (u << 8)) * ldy + ycol0 + col;
        if constexpr (OH) ((_Float16*)Yv)[idx] = (_Float16)(x[u] * inv);
        else              ((float*)Yv)[idx] = x[u] * inv;
    }
}

__global__ __launch_bounds__(256)
void gh_log(const float* __restrict__ H, float* __restrict__ logH, int n)
{
    int i = blockIdx.x * 256 + threadIdx.x;
    if (i < n) logH[i] = __logf(H[i]);
}

// kl[i,j] = sum_n H[n,i]*(logH[n,i] - log(0.5*(H[n,i]+H[n,j]))) ; one wave/(i,j)
__global__ __launch_bounds__(32)
void gh_kl(const float* __restrict__ H, const float* __restrict__ logH,
           float* __restrict__ kl)
{
    const int j = blockIdx.x, i = blockIdx.y, lane = threadIdx.x;
    float acc = 0.f;
    for (int n = lane; n < 1024; n += 32) {
        float hi = H[(size_t)n * 256 + i];
        float hj = H[(size_t)n * 256 + j];
        acc += hi * (logH[(size_t)n * 256 + i] - __logf(0.5f * (hi + hj)));
    }
    for (int o = 16; o > 0; o >>= 1) acc += __shfl_down(acc, o, 32);
    if (lane == 0) kl[(size_t)i * 256 + j] = acc;
}

// jm[j] = mean_i 0.5*(kl[i,j]+kl[j,i]); w = softmax((jm-mean)/(std_ddof1+eps))
__global__ __launch_bounds__(256)
void gh_jmw(const float* __restrict__ kl, float* __restrict__ w)
{
    const int j = threadIdx.x;
    __shared__ float sm[256];
    float s = 0.f;
    for (int i = 0; i < 256; ++i)
        s += kl[(size_t)i * 256 + j] + kl[(size_t)j * 256 + i];
    const float jm = s * (0.5f / 256.0f);

    sm[j] = jm; __syncthreads();
    for (int st = 128; st > 0; st >>= 1) { if (j < st) sm[j] += sm[j + st]; __syncthreads(); }
    const float mean = sm[0] * (1.0f / 256.0f); __syncthreads();

    const float d = jm - mean;
    sm[j] = d * d; __syncthreads();
    for (int st = 128; st > 0; st >>= 1) { if (j < st) sm[j] += sm[j + st]; __syncthreads(); }
    const float sd = sqrtf(sm[0] * (1.0f / 255.0f)) + GH_EPS; __syncthreads();

    const float z = d / sd;
    sm[j] = z; __syncthreads();
    for (int st = 128; st > 0; st >>= 1) { if (j < st) sm[j] = fmaxf(sm[j], sm[j + st]); __syncthreads(); }
    const float mx = sm[0]; __syncthreads();

    const float e = __expf(z - mx);
    sm[j] = e; __syncthreads();
    for (int st = 128; st > 0; st >>= 1) { if (j < st) sm[j] += sm[j + st]; __syncthreads(); }
    w[j] = e / sm[0];
}

// ---------------------------------------------------------------------------
extern "C" void kernel_launch(void* const* d_in, const int* in_sizes, int n_in,
                              void* d_out, int out_size, void* d_ws, size_t ws_size,
                              hipStream_t stream)
{
    (void)in_sizes; (void)n_in; (void)out_size; (void)ws_size;

    const float* xt = (const float*)d_in[0];   // [1024, 512]
    const float* xn = (const float*)d_in[1];
    auto tagp = [&](int tag, int idx) { return (const float*)d_in[2 + tag * 8 + idx]; };
    const float* f1w = (const float*)d_in[34];
    const float* f1b = (const float*)d_in[35];
    const float* f2w = (const float*)d_in[36];
    const float* f2b = (const float*)d_in[37];
    const float* theta_t = (const float*)d_in[38];
    const float* theta_n = (const float*)d_in[39];

    // ---- workspace: f32 region then f16 region (~28 MB total) ----
    float* W     = (float*)d_ws;
    float* sbuf  = W + 0;         // 2 x 1024x1024 f32 head scores
    float* mlp2f = W + 2097152;   // 1024x256 f32 (pre-norm_softmax)
    float* logH  = W + 2359296;   // 1024x256
    float* klb   = W + 2621440;   // 256x256
    float* wv    = W + 2686976;   // 256
    float* G     = W + 2687232;   // 256x512
    float* P     = W + 2818304;   // 256x512

    _Float16* Wh    = (_Float16*)(W + 2949376);
    _Float16* xth   = Wh + 0;        // 1024x512
    _Float16* xnh   = Wh + 524288;
    _Float16* qh    = Wh + 1048576;  // 1024x512
    _Float16* kh    = Wh + 1572864;
    _Float16* abufh = Wh + 2097152;  // 1024x1024
    _Float16* hidh  = Wh + 3145728;  // 1024x256
    _Float16* Hcath = Wh + 3407872;  // 1024x1024
    _Float16* wqh   = Wh + 4456448;  // 4 x 512x512
    _Float16* wkh   = Wh + 5505024;  // 4 x 512x512
    _Float16* h1wh  = Wh + 6553600;  // 4 x 256x1024
    _Float16* h2wh  = Wh + 7602176;  // 4 x 256x256
    _Float16* f1wh  = Wh + 7864320;  // 256x1024
    _Float16* f2wh  = Wh + 8126464;  // 256x256

    float* out  = (float*)d_out;     // [xt' | xn' | H]
    float* Hout = out + 1048576;     // 1024x256 f32

    const dim3 blk(32);
    auto CVT = [&](const float* s, _Float16* d, int n) {
        gh_tof16<<<dim3(n / 1024), dim3(256), 0, stream>>>(s, d, n);
    };

    // ---- one-time (per launch) f32 -> f16 operand conversion ----
    CVT(xt, xth, 524288);
    CVT(xn, xnh, 524288);
    for (int tag = 0; tag < 4; ++tag) {
        CVT(tagp(tag, 0), wqh  + tag * 262144, 262144);
        CVT(tagp(tag, 1), wkh  + tag * 262144, 262144);
        CVT(tagp(tag, 4), h1wh + tag * 262144, 262144);
        CVT(tagp(tag, 6), h2wh + tag * 65536,  65536);
    }
    CVT(f1w, f1wh, 262144);
    CVT(f2w, f2wh, 65536);

    const _Float16* qinh[4] = {xth, xnh, xth, xnh};  // t, n, tn, nt
    const _Float16* kinh[4] = {xth, xnh, xnh, xth};

    for (int tag = 0; tag < 4; ++tag) {
        const float* bq  = tagp(tag, 2); const float* bk  = tagp(tag, 3);
        const float* h1b = tagp(tag, 5); const float* h2b = tagp(tag, 7);

        // Q/K projections: x @ W.T + b  (M=1024, N=512, K=512) -> f16
        gh_gemm_h<true,0,true><<<dim3(8, 64), blk, 0, stream>>>(
            qinh[tag], wqh + tag * 262144, bq, qh, 512, 512, 512, 512, 1.f);
        gh_gemm_h<true,0,true><<<dim3(8, 64), blk, 0, stream>>>(
            kinh[tag], wkh + tag * 262144, bk, kh, 512, 512, 512, 512, 1.f);
        // per-head scores: s[h] = (q_h @ k_h^T)/16  (M=N=1024, K=256) -> f32
        for (int h = 0; h < 2; ++h)
            gh_gemm_h<false,0,false><<<dim3(16, 64), blk, 0, stream>>>(
                qh + h * 256, kh + h * 256, nullptr,
                sbuf + (size_t)h * 1048576, 256, 512, 512, 1024, 0.0625f);
        // head-averaged row softmax -> a (f16)
        gh_softmax_rows_avg<<<dim3(1024), dim3(256), 0, stream>>>(sbuf, abufh);
        // incidence MLP
        gh_gemm_h<true,1,true><<<dim3(4, 64), blk, 0, stream>>>(
            abufh, h1wh + tag * 262144, h1b, hidh, 1024, 1024, 1024, 256, 1.f);
        gh_gemm_h<true,0,false><<<dim3(4, 64), blk, 0, stream>>>(
            hidh, h2wh + tag * 65536, h2b, mlp2f, 256, 256, 256, 256, 1.f);
        // column-standardize + softmax into Hcat slice (f16)
        gh_norm_softmax_col<true><<<dim3(256), dim3(256), 0, stream>>>(
            mlp2f, 256, Hcath, 1024, tag * 256);
    }

    // fusion MLP + norm_softmax -> H (f32, straight into d_out)
    gh_gemm_h<true,1,true><<<dim3(4, 64), blk, 0, stream>>>(
        Hcath, f1wh, f1b, hidh, 1024, 1024, 1024, 256, 1.f);
    gh_gemm_h<true,0,false><<<dim3(4, 64), blk, 0, stream>>>(
        hidh, f2wh, f2b, mlp2f, 256, 256, 256, 256, 1.f);
    gh_norm_softmax_col<false><<<dim3(256), dim3(256), 0, stream>>>(
        mlp2f, 256, Hout, 256, 0);

    // JSD column weights
    gh_log<<<dim3(1024), dim3(256), 0, stream>>>(Hout, logH, 262144);
    gh_kl <<<dim3(256, 256), dim3(32), 0, stream>>>(Hout, logH, klb);
    gh_jmw<<<dim3(1), dim3(256), 0, stream>>>(klb, wv);

    // xt' = xt + elu((H*w) @ (H^T @ xt) @ theta_t)
    gh_gemm_f<1,false,false,0><<<dim3(8, 16), blk, 0, stream>>>(
        Hout, xt,      nullptr, nullptr, G,   1024, 256, 512, 512, 1.f);
    gh_gemm_f<0,false,false,0><<<dim3(8, 16), blk, 0, stream>>>(
        G,    theta_t, nullptr, nullptr, P,   512,  512, 512, 512, 1.f);
    gh_gemm_f<0,true,true,2><<<dim3(8, 64), blk, 0, stream>>>(
        Hout, P,       wv,      xt,      out, 256,  256, 512, 512, 1.f);
    // xn' = xn + elu((H*w) @ (H^T @ xn) @ theta_n)
    gh_gemm_f<1,false,false,0><<<dim3(8, 16), blk, 0, stream>>>(
        Hout, xn,      nullptr, nullptr, G,   1024, 256, 512, 512, 1.f);
    gh_gemm_f<0,false,false,0><<<dim3(8, 16), blk, 0, stream>>>(
        G,    theta_n, nullptr, nullptr, P,   512,  512, 512, 512, 1.f);
    gh_gemm_f<0,true,true,2><<<dim3(8, 64), blk, 0, stream>>>(
        Hout, P,       wv,      xn,      out + 524288, 256, 256, 512, 512, 1.f);
}
